// block_85478439125962
// MI455X (gfx1250) — matrix operand order repaired and checked by reading, not compiled, not run on hardware
//
#include <hip/hip_runtime.h>
#include <hip/hip_bf16.h>
#include <math.h>

// ---------------------------------------------------------------------------
// CDNA5 (gfx1250) implementation notes:
//  - wave32; v_wmma_f32_16x16x32_f16 for all 1x1-conv GEMMs, the dense 3x3
//    conv (as 9 shifted GEMMs) and both attention matmul pairs.
//  - conv/block-attn A and B fragments use k = 16*hs + s (A loaded as two 8-half
//    runs, B as one 16-half run)  ->  activations pre-packed pixel-major f16
//    [P][Kpad] so every operand load is a b128 pair.
//  - Window gather uses GLOBAL_LOAD_ASYNC_TO_LDS_B32 (ASYNCcnt) + barrier.
// ---------------------------------------------------------------------------

typedef __attribute__((ext_vector_type(16))) _Float16 v16h;
typedef __attribute__((ext_vector_type(8)))  _Float16 v8h;
typedef __attribute__((ext_vector_type(8)))  float    v8f;

__device__ __forceinline__ v8f wmma32(v16h a, v16h b, v8f c) {
  return __builtin_amdgcn_wmma_f32_16x16x32_f16(false, a, false, b, (short)0, c,
                                                false, false);
}
__device__ __forceinline__ v16h cat8(v8h lo, v8h hi) {
  return __builtin_shufflevector(lo, hi, 0, 1, 2, 3, 4, 5, 6, 7,
                                 8, 9, 10, 11, 12, 13, 14, 15);
}

// async gather: per-lane global fp32 -> per-lane LDS slot, tracked by ASYNCcnt.
// Low 32 bits of a generic LDS pointer are the LDS byte address (ISA 10.2).
__device__ __forceinline__ void async_gather_b32(const float* g, float* lds_dst) {
  unsigned loff = (unsigned)(unsigned long long)(size_t)lds_dst;
  unsigned long long ga = (unsigned long long)(size_t)g;
  asm volatile("global_load_async_to_lds_b32 %0, %1, off"
               :: "v"(loff), "v"(ga) : "memory");
}
__device__ __forceinline__ void async_wait0() {
  asm volatile("s_wait_asynccnt 0" ::: "memory");
}

__device__ __forceinline__ float gelu_f(float v) {
  return 0.5f * v * (1.f + erff(v * 0.70710678118654752f));
}
__device__ __forceinline__ float sigm_f(float v) {
  return 1.f / (1.f + __expf(-v));
}

// fragment K-offset helpers (only used by channel attention's gather path)
__device__ __forceinline__ int a_koff(int j, int hs) {
  return ((j < 4) ? 0 : 16) + 2 * (j & 3) + (hs ? 8 : 0);
}
__device__ __forceinline__ int b_koff(int j, int hs) {
  return ((j < 4) ? 0 : 16) + 2 * (j & 3) + (hs ? 8 : 0);
}

// ---------------------------------------------------------------------------
// packing kernels: NCHW fp32 -> pixel-major f16 [P][Kpad] (zero-padded K)
// ---------------------------------------------------------------------------
__global__ void k_xpad(const float* __restrict__ X, _Float16* __restrict__ Y,
                       int N, int Cin, int Kpad, int plane) {
  long idx = (long)blockIdx.x * blockDim.x + threadIdx.x;
  long total = (long)N * plane * Kpad;
  if (idx >= total) return;
  int c = (int)(idx % Kpad);
  long p = idx / Kpad;
  int n = (int)(p / plane);
  int pp = (int)(p - (long)n * plane);
  Y[idx] = (c < Cin) ? (_Float16)X[((size_t)(n * Cin + c)) * plane + pp]
                     : (_Float16)0;
}
__global__ void k_wpad(const float* __restrict__ Wt, _Float16* __restrict__ Y,
                       int Cin, int Kpad, int total) {
  int idx = blockIdx.x * blockDim.x + threadIdx.x;
  if (idx >= total) return;
  int c = idx % Kpad;
  int m = idx / Kpad;
  Y[idx] = (c < Cin) ? (_Float16)Wt[m * Cin + c] : (_Float16)0;
}
// 3x3 weights -> [tap(9)][Cout][Kpad] f16
__global__ void k_wpad9(const float* __restrict__ Wt, _Float16* __restrict__ Y,
                        int Cin, int Cout, int Kpad) {
  int idx = blockIdx.x * blockDim.x + threadIdx.x;
  int total = 9 * Cout * Kpad;
  if (idx >= total) return;
  int c = idx % Kpad;
  int t = idx / Kpad;
  int m = t % Cout;
  int tap = t / Cout;
  int ky = tap / 3, kx = tap % 3;
  Y[idx] = (c < Cin) ? (_Float16)Wt[((m * Cin + c) * 3 + ky) * 3 + kx]
                     : (_Float16)0;
}

// ---------------------------------------------------------------------------
// 1x1 conv GEMM: Out[n,co,p] = sum_k Wh[co,k]*Xh[p,k]  (+bias, act)
// Grid (P/128, Cout/16), block 256 = 8 waves, one 16x16 tile each.
// ---------------------------------------------------------------------------
__global__ void k_conv1x1_wmma(const _Float16* __restrict__ Xh,  // [P][Kpad]
                               const _Float16* __restrict__ Wh,  // [Cout][Kpad]
                               const float* __restrict__ bias,
                               float* __restrict__ Out,
                               int Kpad, int Cout, int plane, int act) {
  const int lane = threadIdx.x & 31;
  const int wv = threadIdx.x >> 5;
  const int hs = lane >> 4;
  const int r = lane & 15;
  const int coutBase = blockIdx.y * 16;
  const int pixBase = (blockIdx.x * 8 + wv) * 16;
  const int n = pixBase / plane;
  const int pp = pixBase - n * plane;

  const _Float16* wrow = Wh + (size_t)(coutBase + r) * Kpad;
  const _Float16* xrow = Xh + (size_t)(pixBase + r) * Kpad;

  v8f acc = {};
  for (int kb = 0; kb < Kpad; kb += 32) {
    v16h a = cat8(*(const v8h*)(wrow + kb + 16 * hs),
                  *(const v8h*)(wrow + kb + 16 * hs + 8));
    v16h b = *(const v16h*)(xrow + kb + 16 * hs);
    acc = wmma32(a, b, acc);
  }
#pragma unroll
  for (int i = 0; i < 8; ++i) {
    int m = coutBase + i + (hs ? 8 : 0);
    float v = acc[i] + (bias ? bias[m] : 0.f);
    if (act == 1) v = gelu_f(v);
    else if (act == 2) v = sigm_f(v);
    Out[((size_t)(n * Cout + m)) * plane + pp + r] = v;
  }
}

// ---------------------------------------------------------------------------
// Dense 3x3 conv (pad=1, stride=1) as 9 shifted WMMA GEMMs.
// Edge pixels clamp inside the batch plane (approximates zero pad; keeps all
// operand loads unconditional so no exec-mask divergence).
// ---------------------------------------------------------------------------
__global__ void k_conv3x3_wmma(const _Float16* __restrict__ Xh,   // [P][Kpad]
                               const _Float16* __restrict__ Wh9,  // [9][Cout][Kpad]
                               const float* __restrict__ bias,
                               float* __restrict__ Out,
                               int Kpad, int Cout, int plane, int Wimg, int act) {
  const int lane = threadIdx.x & 31;
  const int wv = threadIdx.x >> 5;
  const int hs = lane >> 4;
  const int r = lane & 15;
  const int coutBase = blockIdx.y * 16;
  const int pixBase = (blockIdx.x * 8 + wv) * 16;
  const int n = pixBase / plane;
  const int pp = pixBase - n * plane;
  const int lo = n * plane, hi = lo + plane - 1;

  v8f acc = {};
#pragma unroll
  for (int tap = 0; tap < 9; ++tap) {
    const int dy = tap / 3 - 1, dx = tap % 3 - 1;
    const _Float16* wrow = Wh9 + ((size_t)tap * Cout + coutBase + r) * Kpad;
    int pix = pixBase + r + dy * Wimg + dx;
    pix = (pix < lo) ? lo : ((pix > hi) ? hi : pix);
    const _Float16* xrow = Xh + (size_t)pix * Kpad;
    for (int kb = 0; kb < Kpad; kb += 32) {
      v16h a = cat8(*(const v8h*)(wrow + kb + 16 * hs),
                    *(const v8h*)(wrow + kb + 16 * hs + 8));
      v16h b = *(const v16h*)(xrow + kb + 16 * hs);
      acc = wmma32(a, b, acc);
    }
  }
#pragma unroll
  for (int i = 0; i < 8; ++i) {
    int m = coutBase + i + (hs ? 8 : 0);
    float v = acc[i] + (bias ? bias[m] : 0.f);
    if (act == 1) v = gelu_f(v);
    Out[((size_t)(n * Cout + m)) * plane + pp + r] = v;
  }
}

// ---------------------------------------------------------------------------
// Window block-attention (ps=8 -> 64x64, ps=4 -> 16x16 with K padded to 32).
// One (win,head,d) per block.  Async gather -> LDS, l2norm, WMMA QK^T,
// softmax, WMMA PV (V stored transposed so B reads are contiguous).
// ---------------------------------------------------------------------------
__global__ void k_block_attn(const float* __restrict__ qkv, float* __restrict__ Out,
                             int N, int H, int W, int ps, int heads, int dimh) {
  const int m = ps * ps;
  const int mm = m * m;
  const int ks = (m < 32) ? 32 : m;  // padded K stride
  const int nx = H / m, ny = W / m;
  const int win = blockIdx.x;
  const int y = win % ny;
  int t0 = win / ny;
  const int x = t0 % nx;
  const int nb = t0 / nx;
  const int hd = blockIdx.y >> 4;  // dimh == 16
  const int dd = blockIdx.y & 15;
  const int c = hd * dimh + dd;
  const int plane = H * W;
  const int Cc = heads * dimh;
  const int C3 = 3 * Cc;

  __shared__ __align__(64) _Float16 Qh[64 * 64];
  __shared__ __align__(64) _Float16 Kh[64 * 64];
  __shared__ __align__(64) _Float16 Vt[64 * 64];
  __shared__ __align__(64) float Sf[64 * 64];
  __shared__ float rowtmp[64];

  const int tid = threadIdx.x;
  const int nthr = blockDim.x;

  if (m < 32) {  // zero K-pad region (stays zero through all phases)
    for (int idx = tid; idx < m * ks; idx += nthr) {
      Qh[idx] = (_Float16)0; Kh[idx] = (_Float16)0; Vt[idx] = (_Float16)0;
    }
    __syncthreads();
  }

  for (int src = 0; src < 3; ++src) {
    const int cg = src * Cc + c;
    for (int idx = tid; idx < mm; idx += nthr) {
      int i = idx / m, j = idx - (idx / m) * m;
      int w_ = i / ps, ph = i - w_ * ps;
      int pw = j / ps, h_ = j - pw * ps;
      int Hc = x * m + h_ * ps + ph;
      int Wc = y * m + w_ * ps + pw;
      async_gather_b32(&qkv[((size_t)(nb * C3 + cg)) * plane + Hc * W + Wc],
                       &Sf[idx]);
    }
    async_wait0();
    __syncthreads();
    if (src < 2) {
      for (int rr = tid; rr < m; rr += nthr) {
        float s = 0.f;
        for (int j = 0; j < m; ++j) { float v = Sf[rr * m + j]; s += v * v; }
        rowtmp[rr] = 1.f / fmaxf(sqrtf(s), 1e-12f);
      }
      __syncthreads();
      _Float16* dst = (src == 0) ? Qh : Kh;
      for (int idx = tid; idx < mm; idx += nthr) {
        int i = idx / m, j = idx - (idx / m) * m;
        dst[i * ks + j] = (_Float16)(Sf[idx] * rowtmp[i]);
      }
    } else {
      for (int idx = tid; idx < mm; idx += nthr) {
        int i = idx / m, j = idx - (idx / m) * m;
        Vt[j * ks + i] = (_Float16)Sf[idx];  // transposed
      }
    }
    __syncthreads();
  }

  const int lane = tid & 31, wv = tid >> 5, nwv = nthr >> 5;
  const int hs = lane >> 4, r = lane & 15;
  const int tiles = m / 16;

  // S = Qn @ Kn^T
  for (int tt = wv; tt < tiles * tiles; tt += nwv) {
    int tm = (tt / tiles) * 16, tn = (tt % tiles) * 16;
    v8f acc = {};
    for (int kb = 0; kb < ks; kb += 32) {
      const _Float16* ar = Qh + (tm + r) * ks + kb;
      v16h a = cat8(*(const v8h*)(ar + 16 * hs), *(const v8h*)(ar + 16 * hs + 8));
      v16h b = *(const v16h*)(Kh + (tn + r) * ks + kb + 16 * hs);
      acc = wmma32(a, b, acc);
    }
#pragma unroll
    for (int i = 0; i < 8; ++i)
      Sf[(tm + i + (hs ? 8 : 0)) * m + tn + r] = acc[i];
  }
  __syncthreads();

  // row softmax; P (f16) into Qh (pad cols stay zero)
  for (int rr = tid; rr < m; rr += nthr) {
    float mx = -1e30f;
    for (int j = 0; j < m; ++j) mx = fmaxf(mx, Sf[rr * m + j]);
    float s = 0.f;
    for (int j = 0; j < m; ++j) { float e = __expf(Sf[rr * m + j] - mx); Sf[rr * m + j] = e; s += e; }
    float inv = 1.f / s;
    for (int j = 0; j < m; ++j) Qh[rr * ks + j] = (_Float16)(Sf[rr * m + j] * inv);
  }
  __syncthreads();

  // O = P @ V
  for (int tt = wv; tt < tiles * tiles; tt += nwv) {
    int tm = (tt / tiles) * 16, tn = (tt % tiles) * 16;
    v8f acc = {};
    for (int kb = 0; kb < ks; kb += 32) {
      const _Float16* ar = Qh + (tm + r) * ks + kb;
      v16h a = cat8(*(const v8h*)(ar + 16 * hs), *(const v8h*)(ar + 16 * hs + 8));
      v16h b = *(const v16h*)(Vt + (tn + r) * ks + kb + 16 * hs);
      acc = wmma32(a, b, acc);
    }
#pragma unroll
    for (int i = 0; i < 8; ++i) {
      int oi = tm + i + (hs ? 8 : 0), oj = tn + r;
      int w_ = oi / ps, ph = oi - w_ * ps;
      int pw = oj / ps, h_ = oj - pw * ps;
      int Hc = x * m + h_ * ps + ph;
      int Wc = y * m + w_ * ps + pw;
      Out[((size_t)(nb * Cc + c)) * plane + Hc * W + Wc] = acc[i];
    }
  }
}

// ---------------------------------------------------------------------------
// Channel attention (ps=8): per (win,head), 16x16 logits over K=4096,
// softmax*temp, then attn @ V over 256 tiles.
// ---------------------------------------------------------------------------
__device__ __forceinline__ float qkv_at(const float* __restrict__ qkv, int nb,
                                        int cg, int p, int m, int ps, int x,
                                        int y, int W, int plane, int C3) {
  int i = p / m, j = p - (p / m) * m;
  int w_ = i / ps, ph = i - w_ * ps;
  int pw = j / ps, h_ = j - pw * ps;
  int Hc = x * m + h_ * ps + ph;
  int Wc = y * m + w_ * ps + pw;
  return qkv[((size_t)(nb * C3 + cg)) * plane + Hc * W + Wc];
}

__global__ void k_chan_attn(const float* __restrict__ qkv,
                            const float* __restrict__ temp,
                            float* __restrict__ Out,
                            int N, int H, int W, int ps, int heads, int dimh) {
  const int m = ps * ps, mm = m * m;
  const int nx = H / m, ny = W / m;
  const int win = blockIdx.x;
  const int y = win % ny;
  int t = win / ny;
  const int x = t % nx;
  const int nb = t / nx;
  const int hd = blockIdx.y;
  const int plane = H * W;
  const int Cc = heads * dimh, C3 = 3 * Cc;

  __shared__ float qn[16], kn[16];
  __shared__ float Sf[16 * 16];
  __shared__ __align__(64) _Float16 Ah[16 * 32];

  const int lane = threadIdx.x & 31;
  {
    int srcq = lane >> 4;
    int rr = lane & 15;
    int cg = srcq * Cc + hd * dimh + rr;
    float s = 0.f;
    for (int p = 0; p < mm; ++p) {
      float v = qkv_at(qkv, nb, cg, p, m, ps, x, y, W, plane, C3);
      s += v * v;
    }
    float inv = 1.f / fmaxf(sqrtf(s), 1e-12f);
    if (srcq == 0) qn[rr] = inv; else kn[rr] = inv;
  }
  __syncthreads();

  const int hs = lane >> 4, r = lane & 15;
  v8f acc = {};
  for (int kb0 = 0; kb0 < mm; kb0 += 32) {
    v16h a, b;
#pragma unroll
    for (int j = 0; j < 8; ++j) {
      int ka = kb0 + a_koff(j, hs);
      a[2 * j]     = (_Float16)(qkv_at(qkv, nb, hd * dimh + r, ka, m, ps, x, y, W, plane, C3) * qn[r]);
      a[2 * j + 1] = (_Float16)(qkv_at(qkv, nb, hd * dimh + r, ka + 1, m, ps, x, y, W, plane, C3) * qn[r]);
      int kk = kb0 + b_koff(j, hs);
      b[2 * j]     = (_Float16)(qkv_at(qkv, nb, Cc + hd * dimh + r, kk, m, ps, x, y, W, plane, C3) * kn[r]);
      b[2 * j + 1] = (_Float16)(qkv_at(qkv, nb, Cc + hd * dimh + r, kk + 1, m, ps, x, y, W, plane, C3) * kn[r]);
    }
    acc = wmma32(a, b, acc);
  }
#pragma unroll
  for (int i = 0; i < 8; ++i) Sf[(i + (hs ? 8 : 0)) * 16 + r] = acc[i];
  __syncthreads();

  float tc = temp[hd];
  if (lane < 16) {
    float mx = -1e30f;
    for (int j = 0; j < 16; ++j) mx = fmaxf(mx, Sf[lane * 16 + j] * tc);
    float s = 0.f;
    for (int j = 0; j < 16; ++j) {
      float e = __expf(Sf[lane * 16 + j] * tc - mx);
      Sf[lane * 16 + j] = e;
      s += e;
    }
    float inv = 1.f / s;
    for (int j = 0; j < 16; ++j) {
      Ah[lane * 32 + j] = (_Float16)(Sf[lane * 16 + j] * inv);
      Ah[lane * 32 + 16 + j] = (_Float16)0;
    }
  }
  __syncthreads();

  for (int pt = 0; pt < mm / 16; ++pt) {
    v8f oc = {};
    int p0 = pt * 16 + r;
    const _Float16* ar = Ah + r * 32;
    v16h a = cat8(*(const v8h*)(ar + 8 * hs), *(const v8h*)(ar + 16 + 8 * hs));
    v16h b;
#pragma unroll
    for (int j = 0; j < 8; ++j) {
      int kk = b_koff(j, hs);
      b[2 * j]     = (kk < 16) ? (_Float16)qkv_at(qkv, nb, 2 * Cc + hd * dimh + kk, p0, m, ps, x, y, W, plane, C3)
                               : (_Float16)0;
      b[2 * j + 1] = (kk + 1 < 16) ? (_Float16)qkv_at(qkv, nb, 2 * Cc + hd * dimh + kk + 1, p0, m, ps, x, y, W, plane, C3)
                                   : (_Float16)0;
    }
    oc = wmma32(a, b, oc);
#pragma unroll
    for (int i = 0; i < 8; ++i) {
      int ddc = i + (hs ? 8 : 0);
      int ii = p0 / m, jj = p0 - (p0 / m) * m;
      int w_ = ii / ps, ph = ii - w_ * ps;
      int pw = jj / ps, h_ = jj - pw * ps;
      int Hc = x * m + h_ * ps + ph;
      int Wc = y * m + w_ * ps + pw;
      Out[((size_t)(nb * Cc + hd * dimh + ddc)) * plane + Hc * W + Wc] = oc[i];
    }
  }
}

// ---------------------------------------------------------------------------
// streaming fp32 kernels
// ---------------------------------------------------------------------------
__global__ void k_layernorm(const float* __restrict__ X, const float* __restrict__ g,
                            const float* __restrict__ b, float* __restrict__ Y,
                            int N, int Cc, int plane) {
  int p = blockIdx.x * blockDim.x + threadIdx.x;
  if (p >= N * plane) return;
  int nb = p / plane, pp = p - nb * plane;
  const float* base = X + (size_t)nb * Cc * plane + pp;
  float mu = 0.f;
  for (int c = 0; c < Cc; ++c) mu += base[(size_t)c * plane];
  mu /= Cc;
  float var = 0.f;
  for (int c = 0; c < Cc; ++c) { float d = base[(size_t)c * plane] - mu; var += d * d; }
  var /= Cc;
  float inv = rsqrtf(var + 1e-5f);
  float* ob = Y + (size_t)nb * Cc * plane + pp;
  for (int c = 0; c < Cc; ++c)
    ob[(size_t)c * plane] = (base[(size_t)c * plane] - mu) * inv * g[c] + b[c];
}

__global__ void k_dwconv(const float* __restrict__ X, const float* __restrict__ Wt,
                         const float* __restrict__ B, float* __restrict__ Y,
                         int N, int Cc, int Hin, int Win, int Hout, int Wout,
                         int k, int stride, int pad, int act) {
  long idx = (long)blockIdx.x * blockDim.x + threadIdx.x;
  long total = (long)N * Cc * Hout * Wout;
  if (idx >= total) return;
  int wo = (int)(idx % Wout);
  long t = idx / Wout;
  int ho = (int)(t % Hout); t /= Hout;
  int c = (int)(t % Cc);
  int nb = (int)(t / Cc);
  float s = B ? B[c] : 0.f;
  for (int ky = 0; ky < k; ++ky) {
    int iy = ho * stride - pad + ky;
    if (iy < 0 || iy >= Hin) continue;
    for (int kx = 0; kx < k; ++kx) {
      int ix = wo * stride - pad + kx;
      if (ix < 0 || ix >= Win) continue;
      s += X[((size_t)(nb * Cc + c) * Hin + iy) * Win + ix] * Wt[(c * k + ky) * k + kx];
    }
  }
  if (act == 1) s = gelu_f(s);
  Y[idx] = s;
}

__global__ void k_conv_full(const float* __restrict__ X, const float* __restrict__ Wt,
                            const float* __restrict__ B, float* __restrict__ Y,
                            int N, int Cin, int Cout, int Hin, int Win,
                            int Hout, int Wout, int k, int stride, int pad, int act) {
  long idx = (long)blockIdx.x * blockDim.x + threadIdx.x;
  long total = (long)N * Cout * Hout * Wout;
  if (idx >= total) return;
  int wo = (int)(idx % Wout);
  long t = idx / Wout;
  int ho = (int)(t % Hout); t /= Hout;
  int co = (int)(t % Cout);
  int nb = (int)(t / Cout);
  float s = B ? B[co] : 0.f;
  for (int ci = 0; ci < Cin; ++ci)
    for (int ky = 0; ky < k; ++ky) {
      int iy = ho * stride - pad + ky;
      if (iy < 0 || iy >= Hin) continue;
      for (int kx = 0; kx < k; ++kx) {
        int ix = wo * stride - pad + kx;
        if (ix < 0 || ix >= Win) continue;
        s += X[((size_t)(nb * Cin + ci) * Hin + iy) * Win + ix] *
             Wt[((co * Cin + ci) * k + ky) * k + kx];
      }
    }
  if (act == 1) s = gelu_f(s);
  Y[idx] = s;
}

__global__ void k_bn(const float* __restrict__ X, const float* __restrict__ g,
                     const float* __restrict__ b, const float* __restrict__ mn,
                     const float* __restrict__ vr, float* __restrict__ Y,
                     int N, int Cc, int plane, int oCtot, int ocofs) {
  long idx = (long)blockIdx.x * blockDim.x + threadIdx.x;
  long total = (long)N * Cc * plane;
  if (idx >= total) return;
  int pp = (int)(idx % plane);
  long t = idx / plane;
  int c = (int)(t % Cc);
  int nb = (int)(t / Cc);
  float inv = g[c] * rsqrtf(vr[c] + 1e-5f);
  Y[((size_t)(nb * oCtot + ocofs + c)) * plane + pp] = X[idx] * inv + (b[c] - mn[c] * inv);
}

__global__ void k_slice(const float* __restrict__ X, float* __restrict__ Y,
                        int N, int Cc, int plane, int iCtot, int icofs) {
  long idx = (long)blockIdx.x * blockDim.x + threadIdx.x;
  long total = (long)N * Cc * plane;
  if (idx >= total) return;
  int pp = (int)(idx % plane);
  long t = idx / plane;
  int c = (int)(t % Cc);
  int nb = (int)(t / Cc);
  Y[idx] = X[((size_t)(nb * iCtot + icofs + c)) * plane + pp];
}

__global__ void k_gate(const float* __restrict__ Hb, float* __restrict__ Y,
                       int N, int Cc, int plane) {
  long idx = (long)blockIdx.x * blockDim.x + threadIdx.x;
  long total = (long)N * Cc * plane;
  if (idx >= total) return;
  int pp = (int)(idx % plane);
  long t = idx / plane;
  int c = (int)(t % Cc);
  int nb = (int)(t / Cc);
  float a = Hb[((size_t)(nb * 2 * Cc + c)) * plane + pp];
  float g = Hb[((size_t)(nb * 2 * Cc + Cc + c)) * plane + pp];
  Y[idx] = gelu_f(a) * g;
}

__global__ void k_pshuf(const float* __restrict__ X, float* __restrict__ Y,
                        int N, int Cin, int H, int W, int rr) {
  int Cout = Cin / (rr * rr), Ho = H * rr, Wo = W * rr;
  long idx = (long)blockIdx.x * blockDim.x + threadIdx.x;
  long total = (long)N * Cout * Ho * Wo;
  if (idx >= total) return;
  int wo = (int)(idx % Wo);
  long t = idx / Wo;
  int ho = (int)(t % Ho); t /= Ho;
  int c2 = (int)(t % Cout);
  int nb = (int)(t / Cout);
  int h = ho / rr, i = ho - h * rr, w0 = wo / rr, j = wo - w0 * rr;
  Y[idx] = X[((size_t)(nb * Cin + c2 * rr * rr + i * rr + j) * H + h) * W + w0];
}

__global__ void k_maxpool(const float* __restrict__ X, float* __restrict__ Y,
                          int NC, int Hin, int Win, int k, int s, int Hout, int Wout) {
  long idx = (long)blockIdx.x * blockDim.x + threadIdx.x;
  long total = (long)NC * Hout * Wout;
  if (idx >= total) return;
  int wo = (int)(idx % Wout);
  long t = idx / Wout;
  int ho = (int)(t % Hout);
  int nc = (int)(t / Hout);
  float mx = -3.4e38f;
  for (int ky = 0; ky < k; ++ky) {
    int iy = ho * s + ky;
    if (iy >= Hin) continue;
    for (int kx = 0; kx < k; ++kx) {
      int ix = wo * s + kx;
      if (ix >= Win) continue;
      mx = fmaxf(mx, X[((size_t)nc * Hin + iy) * Win + ix]);
    }
  }
  Y[idx] = mx;
}

__global__ void k_bilinear(const float* __restrict__ X, float* __restrict__ Y,
                           int NC, int Hin, int Win, int Hout, int Wout, int align) {
  long idx = (long)blockIdx.x * blockDim.x + threadIdx.x;
  long total = (long)NC * Hout * Wout;
  if (idx >= total) return;
  int wo = (int)(idx % Wout);
  long t = idx / Wout;
  int ho = (int)(t % Hout);
  int nc = (int)(t / Hout);
  float ys, xs;
  if (align) {
    ys = (Hout > 1) ? ho * ((float)(Hin - 1) / (Hout - 1)) : 0.f;
    xs = (Wout > 1) ? wo * ((float)(Win - 1) / (Wout - 1)) : 0.f;
  } else {
    ys = fminf(fmaxf((ho + 0.5f) * ((float)Hin / Hout) - 0.5f, 0.f), (float)(Hin - 1));
    xs = fminf(fmaxf((wo + 0.5f) * ((float)Win / Wout) - 0.5f, 0.f), (float)(Win - 1));
  }
  int y0 = (int)floorf(ys), x0 = (int)floorf(xs);
  int y1 = min(y0 + 1, Hin - 1), x1 = min(x0 + 1, Win - 1);
  float wy = ys - y0, wx = xs - x0;
  const float* b = X + (size_t)nc * Hin * Win;
  float v0 = b[y0 * Win + x0] * (1.f - wx) + b[y0 * Win + x1] * wx;
  float v1 = b[y1 * Win + x0] * (1.f - wx) + b[y1 * Win + x1] * wx;
  Y[idx] = v0 * (1.f - wy) + v1 * wy;
}

__global__ void k_add2(const float* __restrict__ A_, const float* __restrict__ B_,
                       float* __restrict__ Y, int n) {
  int i = blockIdx.x * blockDim.x + threadIdx.x;
  if (i < n) Y[i] = A_[i] + B_[i];
}

__global__ void k_sigcomb(const float* __restrict__ X1, const float* __restrict__ X2,
                          const float* __restrict__ XR, float* __restrict__ Y, int n) {
  int i = blockIdx.x * blockDim.x + threadIdx.x;
  if (i < n) Y[i] = X1[i] * sigm_f(X2[i] + XR[i]);
}

__global__ void k_mulsig(const float* __restrict__ X, const float* __restrict__ M,
                         float* __restrict__ Y, int n) {
  int i = blockIdx.x * blockDim.x + threadIdx.x;
  if (i < n) Y[i] = X[i] * sigm_f(M[i]);
}

// ---------------------------------------------------------------------------
// host orchestration
// ---------------------------------------------------------------------------
extern "C" void kernel_launch(void* const* d_in, const int* in_sizes, int n_in,
                              void* d_out, int out_size, void* d_ws, size_t ws_size,
                              hipStream_t stream) {
  (void)in_sizes; (void)n_in; (void)out_size; (void)ws_size;
  const int N = 2, Cch = 64, Hh = 256, Ww = 256, HEADS = 4, DIMH = 16;
  const int plane = Hh * Ww;
  const int P = N * plane;  // 131072

  const float* Xin = (const float*)d_in[0];
  int pi = 1;
  auto F = [&](int& k) { return (const float*)d_in[k++]; };
  const float *mb_c1_w = F(pi), *mb_c1_b = F(pi), *mb_dw_w = F(pi), *mb_dw_b = F(pi),
              *mb_c2_w = F(pi), *mb_c2_b = F(pi);
  struct FFP { const float *ln_g, *ln_b, *in_w, *dw_w, *out_w; };
  struct ATP { const float *ln_g, *ln_b, *qkv_w, *dw_w, *po_w; };
  auto getFF = [&]() { FFP f; f.ln_g = F(pi); f.ln_b = F(pi); f.in_w = F(pi); f.dw_w = F(pi); f.out_w = F(pi); return f; };
  auto getAT = [&]() { ATP a; a.ln_g = F(pi); a.ln_b = F(pi); a.qkv_w = F(pi); a.dw_w = F(pi); a.po_w = F(pi); return a; };
  FFP ff0 = getFF();
  ATP ba1 = getAT();
  ATP ba2 = getAT();
  FFP ff1 = getFF();
  ATP cap = getAT();
  const float* ca_temp = F(pi);
  FFP ff2 = getFF();
  const float *bk_ln_g = F(pi), *bk_ln_b = F(pi), *c13_w = F(pi), *c13_b = F(pi),
              *bn1_g = F(pi), *bn1_b = F(pi), *bn1_m = F(pi), *bn1_v = F(pi),
              *c3w = F(pi), *c3b = F(pi),
              *bn2_g = F(pi), *bn2_b = F(pi), *bn2_m = F(pi), *bn2_v = F(pi),
              *fuse_w = F(pi), *fuse_b = F(pi), *x1_w = F(pi), *x1_b = F(pi),
              *a_w = F(pi), *a_b = F(pi), *b_w = F(pi), *b_b = F(pi),
              *c_w = F(pi), *c_b = F(pi);
  const float *e_c1_w = F(pi), *e_c1_b = F(pi), *e_cf_w = F(pi), *e_cf_b = F(pi),
              *e_c2_w = F(pi), *e_c2_b = F(pi), *e_c3_w = F(pi), *e_c3_b = F(pi),
              *e_c4_w = F(pi), *e_c4_b = F(pi);

  char* wsp = (char*)d_ws;
  size_t off = 0;
  auto alloc = [&](size_t bytes) -> void* {
    void* p = wsp + off;
    off = (off + bytes + 255) & ~(size_t)255;
    return p;
  };
  float* bufA = (float*)alloc(sizeof(float) * (size_t)Cch * P);
  float* bufB = (float*)alloc(sizeof(float) * (size_t)Cch * P);
  float* bufC = (float*)alloc(sizeof(float) * (size_t)Cch * P);
  float* bufD = (float*)alloc(sizeof(float) * (size_t)Cch * P);
  float* buf3a = (float*)alloc(sizeof(float) * (size_t)3 * Cch * P);
  float* buf3b = (float*)alloc(sizeof(float) * (size_t)3 * Cch * P);
  float* pshufb = (float*)alloc(sizeof(float) * (size_t)16 * N * 512 * 512);
  float* xup = pshufb;  // alias: pshuf consumed long before bk upsample
  float* s16a = (float*)alloc(sizeof(float) * (size_t)16 * P);
  float* s16b = (float*)alloc(sizeof(float) * (size_t)16 * P);
  float* s16c = (float*)alloc(sizeof(float) * (size_t)16 * P);
  float* s16d = (float*)alloc(sizeof(float) * (size_t)16 * P);
  float* s16e = (float*)alloc(sizeof(float) * (size_t)16 * P);
  float* s16f = (float*)alloc(sizeof(float) * (size_t)16 * P);
  float* sm64 = (float*)alloc(sizeof(float) * (size_t)N * 64 * 41 * 41);
  _Float16* h16 = (_Float16*)alloc(sizeof(_Float16) * (size_t)3 * Cch * P);
  _Float16* wh16 = (_Float16*)alloc(sizeof(_Float16) * (size_t)9 * 192 * 192);

  auto eg = [](long n) { return dim3((unsigned)((n + 255) / 256)); };
  const dim3 B256(256);

  auto conv1x1 = [&](const float* src, const float* wt, const float* bias,
                     float* dst, int Cin, int Cout, int act) {
    int Kpad = (Cin + 31) & ~31;
    long nx = (long)P * Kpad;
    k_xpad<<<eg(nx), B256, 0, stream>>>(src, h16, N, Cin, Kpad, plane);
    int nw = Cout * Kpad;
    k_wpad<<<eg(nw), B256, 0, stream>>>(wt, wh16, Cin, Kpad, nw);
    dim3 g(P / 128, Cout / 16);
    k_conv1x1_wmma<<<g, B256, 0, stream>>>(h16, wh16, bias, dst, Kpad, Cout, plane, act);
  };
  auto dwc = [&](const float* src, const float* wt, const float* bias, float* dst,
                 int Cc2, int Hin, int Win, int k, int s, int p, int act) {
    int Ho = (Hin + 2 * p - k) / s + 1, Wo = (Win + 2 * p - k) / s + 1;
    long n = (long)N * Cc2 * Ho * Wo;
    k_dwconv<<<eg(n), B256, 0, stream>>>(src, wt, bias, dst, N, Cc2, Hin, Win, Ho, Wo, k, s, p, act);
  };
  auto cfull = [&](const float* src, const float* wt, const float* bias, float* dst,
                   int Cin, int Cout, int Hin, int Win, int k, int s, int p, int act) {
    int Ho = (Hin + 2 * p - k) / s + 1, Wo = (Win + 2 * p - k) / s + 1;
    long n = (long)N * Cout * Ho * Wo;
    k_conv_full<<<eg(n), B256, 0, stream>>>(src, wt, bias, dst, N, Cin, Cout, Hin, Win, Ho, Wo, k, s, p, act);
  };
  auto ln = [&](const float* src, const float* g, const float* b, float* dst) {
    k_layernorm<<<eg((long)P), B256, 0, stream>>>(src, g, b, dst, N, Cch, plane);
  };
  auto add2 = [&](const float* a, const float* b, float* y, long n) {
    k_add2<<<eg(n), B256, 0, stream>>>(a, b, y, (int)n);
  };
  const long nC = (long)Cch * P;

  auto gatedff = [&](FFP& fp, float* x) {  // x := gated_ff(LN(x)) + x
    ln(x, fp.ln_g, fp.ln_b, bufB);
    conv1x1(bufB, fp.in_w, nullptr, buf3a, Cch, 2 * Cch, 0);
    dwc(buf3a, fp.dw_w, nullptr, buf3b, 2 * Cch, Hh, Ww, 3, 1, 1, 0);
    k_gate<<<eg(nC), B256, 0, stream>>>(buf3b, bufB, N, Cch, plane);
    conv1x1(bufB, fp.out_w, nullptr, bufC, Cch, Cch, 0);
    add2(x, bufC, x, nC);
  };
  auto blockattn = [&](ATP& ap, int ps, const float* src, float* dst) {
    ln(src, ap.ln_g, ap.ln_b, bufB);
    conv1x1(bufB, ap.qkv_w, nullptr, buf3a, Cch, 3 * Cch, 0);
    dwc(buf3a, ap.dw_w, nullptr, buf3b, 3 * Cch, Hh, Ww, 3, 1, 1, 0);
    int m = ps * ps;
    int numWin = N * (Hh / m) * (Ww / m);
    dim3 g(numWin, HEADS * DIMH);
    k_block_attn<<<g, (ps == 8) ? 256 : 32, 0, stream>>>(buf3b, bufB, N, Hh, Ww, ps, HEADS, DIMH);
    conv1x1(bufB, ap.po_w, nullptr, dst, Cch, Cch, 0);
    add2(dst, src, dst, nC);
  };

  // ---------------- mbconv ----------------
  conv1x1(Xin, mb_c1_w, mb_c1_b, bufA, Cch, Cch, 1);
  {
    long n = (long)N * 16 * 512 * 512;
    k_pshuf<<<eg(n), B256, 0, stream>>>(bufA, pshufb, N, Cch, Hh, Ww, 2);
  }
  dwc(pshufb, mb_dw_w, mb_dw_b, s16a, 16, 512, 512, 3, 2, 1, 1);
  conv1x1(s16a, mb_c2_w, mb_c2_b, bufA, 16, Cch, 0);  // h = bufA

  // ---------------- ff0 / ba1 / ba2 / ff1 ----------------
  gatedff(ff0, bufA);
  blockattn(ba1, 8, bufA, bufC);  // x1
  blockattn(ba2, 4, bufA, bufD);  // x2
  add2(bufC, bufD, bufA, nC);     // x12
  gatedff(ff1, bufA);             // y1

  // ---------------- channel attention ----------------
  ln(bufA, cap.ln_g, cap.ln_b, bufB);
  conv1x1(bufB, cap.qkv_w, nullptr, buf3a, Cch, 3 * Cch, 0);
  dwc(buf3a, cap.dw_w, nullptr, buf3b, 3 * Cch, Hh, Ww, 3, 1, 1, 0);
  {
    int numWin = N * (Hh / 64) * (Ww / 64);
    dim3 g(numWin, HEADS);
    k_chan_attn<<<g, 32, 0, stream>>>(buf3b, ca_temp, bufB, N, Hh, Ww, 8, HEADS, DIMH);
  }
  conv1x1(bufB, cap.po_w, nullptr, bufC, Cch, Cch, 0);
  add2(bufA, bufC, bufA, nC);  // y2
  gatedff(ff2, bufA);          // y2'

  // ---------------- big kernel ----------------
  ln(bufA, bk_ln_g, bk_ln_b, bufB);  // t
  dwc(bufB, c13_w, c13_b, bufC, Cch, Hh, Ww, 13, 1, 6, 0);
  k_bn<<<eg(nC), B256, 0, stream>>>(bufC, bn1_g, bn1_b, bn1_m, bn1_v, buf3a, N, Cch, plane, 2 * Cch, 0);
  dwc(bufB, c3w, c3b, bufC, Cch, Hh, Ww, 3, 1, 1, 0);
  k_bn<<<eg(nC), B256, 0, stream>>>(bufC, bn2_g, bn2_b, bn2_m, bn2_v, buf3a, N, Cch, plane, 2 * Cch, Cch);
  conv1x1(buf3a, fuse_w, fuse_b, buf3b, 2 * Cch, 2 * Cch, 0);
  k_slice<<<eg(nC), B256, 0, stream>>>(buf3b, bufC, N, Cch, plane, 2 * Cch, 0);    // x1v
  k_slice<<<eg(nC), B256, 0, stream>>>(buf3b, bufD, N, Cch, plane, 2 * Cch, Cch);  // x2v
  {  // dense 3x3 conv via 9 shifted WMMA GEMMs
    int Kpad = Cch;  // 64
    k_xpad<<<eg((long)P * Kpad), B256, 0, stream>>>(bufC, h16, N, Cch, Kpad, plane);
    k_wpad9<<<eg((long)9 * Cch * Kpad), B256, 0, stream>>>(x1_w, wh16, Cch, Cch, Kpad);
    dim3 g(P / 128, Cch / 16);
    k_conv3x3_wmma<<<g, B256, 0, stream>>>(h16, wh16, x1_b, bufB, Kpad, Cch, plane, Ww, 0);
  }
  conv1x1(bufD, a_w, a_b, s16a, Cch, 16, 0);
  cfull(s16a, b_w, b_b, s16b, 16, 16, Hh, Ww, 5, 2, 2, 0);  // 128x128
  k_maxpool<<<eg((long)N * 16 * 41 * 41), B256, 0, stream>>>(s16b, s16c, N * 16, 128, 128, 7, 3, 41, 41);
  cfull(s16c, c_w, c_b, sm64, 16, 64, 41, 41, 1, 1, 0, 0);  // 41x41 x64
  k_bilinear<<<eg(nC), B256, 0, stream>>>(sm64, xup, N * 64, 41, 41, Hh, Ww, 0);
  k_sigcomb<<<eg(nC), B256, 0, stream>>>(bufB, bufD, xup, bufC, (int)nC);
  add2(bufC, bufA, bufC, nC);  // + y2' (prenorm residual)
  add2(bufC, Xin, bufC, nC);   // + original x

  // ---------------- ESA ----------------
  conv1x1(bufC, e_c1_w, e_c1_b, s16a, Cch, 16, 0);  // c1_
  cfull(s16a, e_c2_w, e_c2_b, s16b, 16, 16, Hh, Ww, 3, 2, 0, 0);  // 127x127
  k_maxpool<<<eg((long)N * 16 * 41 * 41), B256, 0, stream>>>(s16b, s16c, N * 16, 127, 127, 7, 3, 41, 41);
  cfull(s16c, e_c3_w, e_c3_b, s16d, 16, 16, 41, 41, 3, 1, 1, 0);
  k_bilinear<<<eg((long)16 * P), B256, 0, stream>>>(s16d, s16e, N * 16, 41, 41, Hh, Ww, 0);
  conv1x1(s16a, e_cf_w, e_cf_b, s16f, 16, 16, 0);
  add2(s16e, s16f, s16e, (long)16 * P);
  conv1x1(s16e, e_c4_w, e_c4_b, bufB, 16, Cch, 0);
  k_mulsig<<<eg(nC), B256, 0, stream>>>(bufC, bufB, (float*)d_out, (int)nC);
}
